// SelfAttentionBlock_88562225643976
// MI455X (gfx1250) — matrix operand order repaired and checked by reading, not compiled, not run on hardware
//
#include <hip/hip_runtime.h>

// ---------------------------------------------------------------------------
// SelfAttentionBlock for MI455X (gfx1250, wave32, WMMA + TDM).
// Pipeline (bf16 WMMA, f32 accumulate; ~80MB of intermediates stay in 192MB L2):
//   cvt  : fp32 weights -> bf16
//   gn   : GroupNorm -> bf16 xn
//   gemm : qkv = W_qkv @ xn (+bias), q pre-scaled by 1/sqrt(64)
//   attn : flash-style attention per (b, head, 128-row t block)
//   gemm : out = W_proj @ hout + bias + x   (fp32 output)
// W tiles (gemm) and V tiles (attn) are plain 2D strided tiles -> staged by
// the Tensor Data Mover; K/X tiles need a transpose so they are staged
// manually (vectorized). Softmax row reductions use DPP16 butterflies
// (VALU, co-executes with WMMA) instead of ds_bpermute.
// ---------------------------------------------------------------------------

typedef __attribute__((ext_vector_type(16))) __bf16 v16bf;
typedef __attribute__((ext_vector_type(8)))  float  v8f;

#define TT   1024   // tokens (32*32)
#define CC   512
#define BB   16
#define NH   8
#define DH   64

#if __has_builtin(__builtin_amdgcn_tensor_load_to_lds)
#define USE_TDM 1
#else
#define USE_TDM 0
#endif

#if __has_builtin(__builtin_amdgcn_update_dpp)
#define USE_DPP 1
#else
#define USE_DPP 0
#endif

__device__ inline v8f wmma_bf16(v16bf a, v16bf b, v8f c) {
  return __builtin_amdgcn_wmma_f32_16x16x32_bf16(
      false, a, false, b, (short)0, c, false, false);
}

// A fragment: 16x32 (MxK) bf16, element (m,k) at base[m*ld + k].
__device__ inline v16bf load_frag_a(const __bf16* base, int ld) {
  int lane = threadIdx.x & 31;
  int m  = lane & 15;
  int kh = (lane >> 4) & 1;
  const __bf16* p = base + m * ld + kh * 8;
  v16bf a;
#pragma unroll
  for (int i = 0; i < 8; ++i) { a[i] = p[i]; a[8 + i] = p[16 + i]; }
  return a;
}

// B fragment: 32x16 (KxN) bf16 from transposed tile Bt[n*ld + k].
__device__ inline v16bf load_frag_b(const __bf16* base, int ld) {
  int lane = threadIdx.x & 31;
  int n  = lane & 15;
  int kh = (lane >> 4) & 1;
  const __bf16* p = base + n * ld + kh * 8;
  v16bf b;
#pragma unroll
  for (int i = 0; i < 8; ++i) { b[i] = p[i]; b[8 + i] = p[16 + i]; }
  return b;
}

// ---- 16-lane all-reduce (max / sum) -----------------------------------
#if USE_DPP
template <int CTRL>
__device__ inline float dpp_perm(float x) {
  int r = __builtin_amdgcn_update_dpp(
      0, __float_as_int(x), CTRL, 0xf, 0xf, true);
  return __int_as_float(r);
}
__device__ inline float red16_max(float x) {
  x = fmaxf(x, dpp_perm<0xB1>(x));    // quad_perm(1,0,3,2)  : xor1
  x = fmaxf(x, dpp_perm<0x4E>(x));    // quad_perm(2,3,0,1)  : xor2
  x = fmaxf(x, dpp_perm<0x141>(x));   // row_half_mirror     : cross-quad
  x = fmaxf(x, dpp_perm<0x140>(x));   // row_mirror          : cross-8
  return x;
}
__device__ inline float red16_sum(float x) {
  x += dpp_perm<0xB1>(x);
  x += dpp_perm<0x4E>(x);
  x += dpp_perm<0x141>(x);
  x += dpp_perm<0x140>(x);
  return x;
}
#else
__device__ inline float red16_max(float x) {
#pragma unroll
  for (int off = 1; off < 16; off <<= 1) x = fmaxf(x, __shfl_xor(x, off, 32));
  return x;
}
__device__ inline float red16_sum(float x) {
#pragma unroll
  for (int off = 1; off < 16; off <<= 1) x += __shfl_xor(x, off, 32);
  return x;
}
#endif

__device__ inline void tensor_wait0() {
#if __has_builtin(__builtin_amdgcn_s_wait_tensorcnt)
  __builtin_amdgcn_s_wait_tensorcnt(0);
#else
  asm volatile("s_wait_tensorcnt 0x0" ::: "memory");
#endif
}

#if USE_TDM
typedef unsigned int v4u  __attribute__((ext_vector_type(4)));
typedef int          v8i_ __attribute__((ext_vector_type(8)));
typedef int          v4i_ __attribute__((ext_vector_type(4)));

// 2D tile load, 2-byte elements: tile_d0 contiguous elems per row, tile_d1
// rows, row stride `stride_el` elements. lds_addr = byte offset in LDS.
__device__ inline void tdm_load_2d(unsigned lds_addr, const void* gaddr,
                                   unsigned tensor_d0, unsigned tensor_d1,
                                   unsigned long long stride_el,
                                   unsigned tile_d0, unsigned tile_d1) {
  unsigned long long ga = (unsigned long long)(uintptr_t)gaddr;
  v4u g0;
  g0[0] = 1u;                                        // count=1, user mode
  g0[1] = lds_addr;                                  // lds_addr [63:32]
  g0[2] = (unsigned)(ga & 0xffffffffu);              // global_addr lo
  g0[3] = (unsigned)((ga >> 32) & 0x01ffffffu) | (2u << 30);  // hi | type=2
  v8i_ g1;
  g1[0] = (int)(1u << 16);                           // data_size=1 (2 bytes)
  g1[1] = (int)((tensor_d0 & 0xffffu) << 16);        // tensor_dim0 lo16
  g1[2] = (int)(((tensor_d0 >> 16) & 0xffffu) | ((tensor_d1 & 0xffffu) << 16));
  g1[3] = (int)(((tensor_d1 >> 16) & 0xffffu) | ((tile_d0 & 0xffffu) << 16));
  g1[4] = (int)(tile_d1 & 0xffffu);                  // tile_dim1, tile_dim2=0
  g1[5] = (int)(unsigned)(stride_el & 0xffffffffu);  // dim0 stride lo32
  g1[6] = (int)(unsigned)((stride_el >> 32) & 0xffffu);
  g1[7] = 0;
  v4i_ z4 = {0, 0, 0, 0};
  v8i_ z8 = {0, 0, 0, 0, 0, 0, 0, 0};
  __builtin_amdgcn_tensor_load_to_lds(g0, g1, z4, z4, z8, 0);
}

__device__ inline unsigned lds_byte_off(const void* p) {
  // generic LDS pointer: addr[31:0] is the in-group LDS byte offset
  return (unsigned)(uintptr_t)p;
}
#endif

// ---------------------------------------------------------------------------
__global__ __launch_bounds__(256) void cvt_kernel(
    const float* __restrict__ qkvw, const float* __restrict__ projw,
    __bf16* __restrict__ qw, __bf16* __restrict__ pw) {
  int i = blockIdx.x * 256 + threadIdx.x;
  if (i < 3 * CC * CC) qw[i] = (__bf16)qkvw[i];
  if (i < CC * CC)     pw[i] = (__bf16)projw[i];
}

// ---------------------------------------------------------------------------
__global__ __launch_bounds__(256) void gn_kernel(
    const float* __restrict__ x, const float* __restrict__ gs,
    const float* __restrict__ gb, __bf16* __restrict__ xn) {
  __shared__ float red[2][8];
  int tid = threadIdx.x, wave = tid >> 5, lane = tid & 31;
  int b = blockIdx.x >> 5, g = blockIdx.x & 31;
  const float* xp = x + ((size_t)(b * CC + g * 16)) * TT;
  float s = 0.f, ss = 0.f;
  for (int i = tid; i < 16 * TT; i += 256) {
    float v = xp[i]; s += v; ss += v * v;
  }
#pragma unroll
  for (int off = 1; off < 32; off <<= 1) {
    s += __shfl_xor(s, off, 32); ss += __shfl_xor(ss, off, 32);
  }
  if (lane == 0) { red[0][wave] = s; red[1][wave] = ss; }
  __syncthreads();
  if (wave == 0) {
    float a  = (lane < 8) ? red[0][lane] : 0.f;
    float c2 = (lane < 8) ? red[1][lane] : 0.f;
#pragma unroll
    for (int off = 1; off < 8; off <<= 1) {
      a += __shfl_xor(a, off, 32); c2 += __shfl_xor(c2, off, 32);
    }
    if (lane == 0) { red[0][0] = a; red[1][0] = c2; }
  }
  __syncthreads();
  float inv_n = 1.f / (16.f * TT);
  float mu   = red[0][0] * inv_n;
  float var  = red[1][0] * inv_n - mu * mu;
  float rstd = rsqrtf(var + 1e-5f);
  __bf16* op = xn + ((size_t)(b * CC + g * 16)) * TT;
  for (int i = tid; i < 16 * TT; i += 256) {
    int c = g * 16 + (i >> 10);
    float v = (xp[i] - mu) * rstd * gs[c] + gb[c];
    op[i] = (__bf16)v;
  }
}

// ---------------------------------------------------------------------------
// out[b, o, t] = sum_c W[o,c]*X[b,c,t] + bias[o]  (+resid for fp32 output)
// Block tile M=128 (o) x N=64 (t); wave owns a 16-row o strip x 64 t.
template <int MTOT, int NUMOT, bool QSCALE, bool BF16OUT>
__global__ __launch_bounds__(256) void gemm_kernel(
    const __bf16* __restrict__ W, const __bf16* __restrict__ X,
    const float* __restrict__ bias, const float* __restrict__ resid,
    float* __restrict__ outf, __bf16* __restrict__ outbf) {
  __shared__ union {
    struct { __bf16 Wt[128 * 32]; __bf16 Bt[64 * 32]; } s;  // K-step tiles
    __bf16 obf[128 * 64];                                   // bf16 out stage
    float  of32[128 * 64];                                  // f32 out stage
  } sh;
  int tid = threadIdx.x, wave = tid >> 5, lane = tid & 31;
  int blk = blockIdx.x;
  int tt = blk & 15;
  int ot = (blk >> 4) % NUMOT;
  int b  = (blk >> 4) / NUMOT;
  int oBase = ot * 128;
  int t0 = tt * 64;
  int mb = (lane >> 4) * 8;
  int nn = lane & 15;
  int oW = oBase + wave * 16;

  float bv[8];
#pragma unroll
  for (int r = 0; r < 8; ++r) bv[r] = bias[oW + mb + r];
  v8f acc[4];
#pragma unroll
  for (int j = 0; j < 4; ++j)
#pragma unroll
    for (int r = 0; r < 8; ++r) acc[j][r] = bv[r];

  const __bf16* Xb = X + (size_t)b * CC * TT;

  for (int k0 = 0; k0 < CC; k0 += 32) {
    __syncthreads();
#if USE_TDM
    if (wave == 0) {
      tdm_load_2d(lds_byte_off(sh.s.Wt), W + (size_t)oBase * CC + k0,
                  CC, MTOT, CC, 32, 128);
      tensor_wait0();
    }
#else
    // Wt[mo][kk]: 8 contiguous bf16 per iter (b128)
    for (int i = tid; i < 512; i += 256) {
      int mo = i >> 2, k8 = (i & 3) * 8;
      *(uint4*)&sh.s.Wt[mo * 32 + k8] =
          *(const uint4*)&W[(size_t)(oBase + mo) * CC + k0 + k8];
    }
#endif
    // Bt[n][kk] (transposed): read 4 contiguous t, scatter to LDS
    for (int i = tid; i < 512; i += 256) {
      int kk = i >> 4, n4 = (i & 15) * 4;
      union { uint2 u; __bf16 h[4]; } tmp;
      tmp.u = *(const uint2*)&Xb[(size_t)(k0 + kk) * TT + t0 + n4];
#pragma unroll
      for (int u = 0; u < 4; ++u) sh.s.Bt[(n4 + u) * 32 + kk] = tmp.h[u];
    }
    int kn = k0 + 32;
    if (kn < CC) {
      __builtin_prefetch(&W[(size_t)(oBase + (tid >> 1)) * CC + kn + (tid & 1) * 16], 0, 1);
      __builtin_prefetch(&Xb[(size_t)(kn + (tid >> 3)) * TT + t0 + (tid & 7) * 8], 0, 1);
    }
    __syncthreads();
    v16bf af = load_frag_a(&sh.s.Wt[wave * 16 * 32], 32);
#pragma unroll
    for (int j = 0; j < 4; ++j) {
      v16bf bfg = load_frag_b(&sh.s.Bt[j * 16 * 32], 32);
      acc[j] = wmma_bf16(af, bfg, acc[j]);
    }
  }

  // ---- epilogue: stage tile in LDS, then coalesced global writes ----
  __syncthreads();
#pragma unroll
  for (int j = 0; j < 4; ++j)
#pragma unroll
    for (int r = 0; r < 8; ++r) {
      int orow = wave * 16 + mb + r;
      int ncol = j * 16 + nn;
      float v = acc[j][r];
      if (QSCALE) { if (oBase + orow < CC) v *= 0.125f; }  // q * 1/sqrt(64)
      if (BF16OUT) sh.obf[orow * 64 + ncol] = (__bf16)v;
      else         sh.of32[orow * 64 + ncol] = v;
    }
  __syncthreads();
  if (BF16OUT) {
    for (int i = tid * 8; i < 128 * 64; i += 256 * 8) {
      int o = i >> 6, n = i & 63;
      size_t idx = ((size_t)b * MTOT + oBase + o) * TT + t0 + n;
      *(uint4*)&outbf[idx] = *(const uint4*)&sh.obf[i];
    }
  } else {
    for (int i = tid * 4; i < 128 * 64; i += 256 * 4) {
      int o = i >> 6, n = i & 63;
      size_t idx = ((size_t)b * MTOT + oBase + o) * TT + t0 + n;
      float4 rv = *(const float4*)&resid[idx];
      float4 av = *(const float4*)&sh.of32[i];
      av.x += rv.x; av.y += rv.y; av.z += rv.z; av.w += rv.w;
      *(float4*)&outf[idx] = av;
    }
  }
}

// ---------------------------------------------------------------------------
// Flash attention: one block per (b, head, 128 t rows); wave owns 16 t rows.
__global__ __launch_bounds__(256) void attn_kernel(
    const __bf16* __restrict__ qkv, __bf16* __restrict__ hout) {
  __shared__ __bf16 Qt[128 * DH];       // [t_local][d]
  __shared__ __bf16 Kt[32 * DH];        // [s_local][d]   (transposed K)
  __shared__ __bf16 Vs[DH * 32];        // [d][s_local]   (natural V tile)
  __shared__ __bf16 Pst[8][16 * 32];    // per-wave P staging [m][k]

  int tid = threadIdx.x, wave = tid >> 5, lane = tid & 31;
  int blk = blockIdx.x;
  int tblk = blk & 7;
  int h = (blk >> 3) & 7;
  int b = blk >> 6;
  int t0 = tblk * 128;
  int mb = (lane >> 4) * 8;
  int nn = lane & 15;

  const __bf16* qb = qkv + ((size_t)(b * 3 * CC +          h * DH)) * TT;
  const __bf16* kb = qkv + ((size_t)(b * 3 * CC + CC +     h * DH)) * TT;
  const __bf16* vb = qkv + ((size_t)(b * 3 * CC + 2 * CC + h * DH)) * TT;

  // stage Q transposed: read 8 contiguous t (b128), scatter to LDS
  for (int i = tid; i < 1024; i += 256) {
    int dd = i >> 4, t8 = (i & 15) * 8;
    union { uint4 u; __bf16 h8[8]; } tmp;
    tmp.u = *(const uint4*)&qb[(size_t)dd * TT + t0 + t8];
#pragma unroll
    for (int u = 0; u < 8; ++u) Qt[(t8 + u) * DH + dd] = tmp.h8[u];
  }
  __syncthreads();

  v16bf qa0 = load_frag_a(&Qt[wave * 16 * DH], DH);       // K = d 0..31
  v16bf qa1 = load_frag_a(&Qt[wave * 16 * DH + 32], DH);  // K = d 32..63

  float m[8], l[8];
  v8f O[4];
#pragma unroll
  for (int r = 0; r < 8; ++r) { m[r] = -3.0e38f; l[r] = 0.f; }
#pragma unroll
  for (int j = 0; j < 4; ++j)
#pragma unroll
    for (int r = 0; r < 8; ++r) O[j][r] = 0.f;

  for (int s0 = 0; s0 < TT; s0 += 32) {
    __syncthreads();
#if USE_TDM
    if (wave == 0) {
      tdm_load_2d(lds_byte_off(Vs), vb + s0, TT, DH, TT, 32, DH);
      tensor_wait0();
    }
    for (int i = tid; i < 512; i += 256) {          // Kt transpose only
      int dd = i >> 3, s4 = (i & 7) * 4;
      union { uint2 u; __bf16 h4[4]; } tk;
      tk.u = *(const uint2*)&kb[(size_t)dd * TT + s0 + s4];
#pragma unroll
      for (int u = 0; u < 4; ++u) Kt[(s4 + u) * DH + dd] = tk.h4[u];
    }
#else
    for (int i = tid; i < 512; i += 256) {
      int dd = i >> 3, s4 = (i & 7) * 4;
      union { uint2 u; __bf16 h4[4]; } tk, tv;
      tk.u = *(const uint2*)&kb[(size_t)dd * TT + s0 + s4];
      tv.u = *(const uint2*)&vb[(size_t)dd * TT + s0 + s4];
#pragma unroll
      for (int u = 0; u < 4; ++u) Kt[(s4 + u) * DH + dd] = tk.h4[u];
      *(uint2*)&Vs[dd * 32 + s4] = tv.u;
    }
#endif
    __syncthreads();

    // S = Q^T K : 16 t x 32 s (2 col tiles x 2 K-steps over d)
    v8f S0, S1;
#pragma unroll
    for (int r = 0; r < 8; ++r) { S0[r] = 0.f; S1[r] = 0.f; }
    {
      v16bf k00 = load_frag_b(&Kt[0], DH);
      v16bf k01 = load_frag_b(&Kt[32], DH);
      v16bf k10 = load_frag_b(&Kt[16 * DH], DH);
      v16bf k11 = load_frag_b(&Kt[16 * DH + 32], DH);
      S0 = wmma_bf16(qa0, k00, S0);
      S0 = wmma_bf16(qa1, k01, S0);
      S1 = wmma_bf16(qa0, k10, S1);
      S1 = wmma_bf16(qa1, k11, S1);
    }

    // online softmax: 16-lane DPP all-reduce per row (row = mb+r)
    float p0[8], p1[8], alpha[8];
#pragma unroll
    for (int r = 0; r < 8; ++r) {
      float mx = red16_max(fmaxf(S0[r], S1[r]));
      float mn = fmaxf(m[r], mx);
      alpha[r] = __expf(m[r] - mn);
      m[r] = mn;
      p0[r] = __expf(S0[r] - mn);
      p1[r] = __expf(S1[r] - mn);
      float sm = red16_sum(p0[r] + p1[r]);
      l[r] = l[r] * alpha[r] + sm;
    }
#pragma unroll
    for (int j = 0; j < 4; ++j)
#pragma unroll
      for (int r = 0; r < 8; ++r) O[j][r] *= alpha[r];

    // P: D-layout -> per-wave LDS -> A fragment
    __bf16* pw = &Pst[wave][0];
#pragma unroll
    for (int r = 0; r < 8; ++r) {
      pw[(mb + r) * 32 + nn]      = (__bf16)p0[r];
      pw[(mb + r) * 32 + 16 + nn] = (__bf16)p1[r];
    }
    v16bf pa = load_frag_a(pw, 32);
#pragma unroll
    for (int j = 0; j < 4; ++j) {
      v16bf vbf = load_frag_b(&Vs[j * 16 * 32], 32);
      O[j] = wmma_bf16(pa, vbf, O[j]);
    }
  }

  // epilogue: /l, pack 8 bf16, 16B stores contiguous in t
  float inv[8];
#pragma unroll
  for (int r = 0; r < 8; ++r) inv[r] = 1.f / l[r];
#pragma unroll
  for (int j = 0; j < 4; ++j) {
    int ch = h * DH + j * 16 + nn;
    __bf16* dst = hout + ((size_t)(b * CC + ch)) * TT + t0 + wave * 16 + mb;
    __bf16 tmp[8];
#pragma unroll
    for (int r = 0; r < 8; ++r) tmp[r] = (__bf16)(O[j][r] * inv[r]);
    __builtin_memcpy(dst, tmp, 16);
  }
}

// ---------------------------------------------------------------------------
extern "C" void kernel_launch(void* const* d_in, const int* in_sizes, int n_in,
                              void* d_out, int out_size, void* d_ws, size_t ws_size,
                              hipStream_t stream) {
  const float* x        = (const float*)d_in[0];
  const float* gn_scale = (const float*)d_in[1];
  const float* gn_bias  = (const float*)d_in[2];
  const float* qkv_w    = (const float*)d_in[3];
  const float* qkv_b    = (const float*)d_in[4];
  const float* proj_w   = (const float*)d_in[5];
  const float* proj_b   = (const float*)d_in[6];
  float* out = (float*)d_out;

  char* ws = (char*)d_ws;
  __bf16* qkvw_bf  = (__bf16*)(ws);
  __bf16* projw_bf = (__bf16*)(ws + (size_t)1536 * 1024);
  __bf16* xn_bf    = (__bf16*)(ws + (size_t)2 * 1024 * 1024);
  __bf16* qkv_bf   = (__bf16*)(ws + (size_t)18 * 1024 * 1024);
  __bf16* hout_bf  = (__bf16*)(ws + (size_t)66 * 1024 * 1024);

  cvt_kernel<<<3072, 256, 0, stream>>>(qkv_w, proj_w, qkvw_bf, projw_bf);
  gn_kernel<<<BB * 32, 256, 0, stream>>>(x, gn_scale, gn_bias, xn_bf);
  gemm_kernel<3 * CC, 12, true, true><<<BB * 12 * 16, 256, 0, stream>>>(
      qkvw_bf, xn_bf, qkv_b, nullptr, nullptr, qkv_bf);
  attn_kernel<<<BB * NH * (TT / 128), 256, 0, stream>>>(qkv_bf, hout_bf);
  gemm_kernel<CC, 4, false, false><<<BB * 4 * 16, 256, 0, stream>>>(
      projw_bf, hout_bf, proj_b, x, out, nullptr);
}
